// HeposCrossAttention_16157666968224
// MI455X (gfx1250) — matrix operand order repaired and checked by reading, not compiled, not run on hardware
//
#include <hip/hip_runtime.h>

// ---------------- types ----------------
typedef __bf16 bf16;
typedef __bf16 v16bf __attribute__((ext_vector_type(16)));
typedef __bf16 v8bf  __attribute__((ext_vector_type(8)));
typedef __bf16 v4bf  __attribute__((ext_vector_type(4)));
typedef float  v8f   __attribute__((ext_vector_type(8)));
typedef float  v4f   __attribute__((ext_vector_type(4)));

#define T_LEN 1024
#define B_SZ  4
#define S_LEN 8192
#define D_DIM 1024
#define H_NUM 16
#define HD    64
#define NKV   512
#define STRIDE 16

// ---------------- WMMA helpers (ISA 7.12.2 layouts) ----------------
__device__ __forceinline__ v8f zero8() {
  v8f z;
#pragma unroll
  for (int i = 0; i < 8; ++i) z[i] = 0.f;
  return z;
}

// A-fragment 16x32 bf16: lane<16 -> row=lane, K {0..7, 16..23};
//                        lane>=16 -> row=lane-16, K {8..15, 24..31}
__device__ __forceinline__ v16bf frag_a_lds(const bf16* base, int lda, int lane) {
  int row = lane & 15;
  int kb  = (lane < 16) ? 0 : 8;
  const bf16* p = base + row * lda + kb;
  v8bf lo = *(const v8bf*)(p);
  v8bf hi = *(const v8bf*)(p + 16);
  v16bf r;
#pragma unroll
  for (int i = 0; i < 8; ++i) { r[i] = lo[i]; r[i + 8] = hi[i]; }
  return r;
}

// B-fragment 32x16 bf16 (stored as Bs[n][k]): lane n=lane&15,
// K {0..7, 16..23} for lane<16; K {8..15, 24..31} for lane>=16.
__device__ __forceinline__ v16bf frag_b_lds(const bf16* base, int ldb, int lane) {
  int n  = lane & 15;
  int kb = (lane < 16) ? 0 : 8;
  const bf16* p = base + n * ldb + kb;
  v8bf lo = *(const v8bf*)(p);
  v8bf hi = *(const v8bf*)(p + 16);
  v16bf r;
#pragma unroll
  for (int i = 0; i < 8; ++i) { r[i] = lo[i]; r[i + 8] = hi[i]; }
  return r;
}

__device__ __forceinline__ v8f wmma_bf16(v16bf a, v16bf b, v8f c) {
  return __builtin_amdgcn_wmma_f32_16x16x32_bf16(false, a, false, b, (short)0, c, false, false);
}

// async 16B copy global -> LDS (bf16 tiles already converted); ASYNCcnt-tracked
__device__ __forceinline__ void async_cp16(bf16* lds_dst, const bf16* gsrc) {
  unsigned lds_off = (unsigned)(size_t)lds_dst;  // low 32 bits of generic = LDS addr
  asm volatile("global_load_async_to_lds_b128 %0, %1, off"
               :: "v"(lds_off), "v"(gsrc) : "memory");
}
__device__ __forceinline__ void async_wait0() {
  asm volatile("s_wait_asynccnt 0x0" ::: "memory");
}

// ============ Kernel 1: Q projection ============
// C[M=4096,N=1024] = query[m,k] * Wq[n,k]^T (+bq), scatter to q_bf[b][h][t][d]
__global__ void __launch_bounds__(256) proj_q_kernel(
    const float* __restrict__ X, const float* __restrict__ W,
    const float* __restrict__ bias, unsigned short* __restrict__ outp) {
  __shared__ __align__(16) bf16 As[128 * 40];
  __shared__ __align__(16) bf16 Bs[128 * 40];
  const int tid = threadIdx.x, lane = tid & 31, wav = tid >> 5;
  const int mBase = blockIdx.x * 128;
  const int nBase = blockIdx.y * 128;
  v8f acc[8];
#pragma unroll
  for (int i = 0; i < 8; ++i) acc[i] = zero8();

  const int srow = tid >> 3, scol = (tid & 7) * 4;   // staging coords (row += 32/iter)

  for (int kb = 0; kb < D_DIM; kb += 32) {
    __syncthreads();
    v4f fa[4], fb[4];
#pragma unroll
    for (int it = 0; it < 4; ++it) {   // issue all 8 global loads, then one wait
      int row = srow + it * 32;
      fa[it] = *(const v4f*)(X + (size_t)(mBase + row) * D_DIM + kb + scol);
      fb[it] = *(const v4f*)(W + (size_t)(nBase + row) * D_DIM + kb + scol);
    }
#pragma unroll
    for (int it = 0; it < 4; ++it) {
      int row = srow + it * 32;
      v4bf ha; ha[0]=(bf16)fa[it][0]; ha[1]=(bf16)fa[it][1]; ha[2]=(bf16)fa[it][2]; ha[3]=(bf16)fa[it][3];
      v4bf hb; hb[0]=(bf16)fb[it][0]; hb[1]=(bf16)fb[it][1]; hb[2]=(bf16)fb[it][2]; hb[3]=(bf16)fb[it][3];
      *(v4bf*)(As + row * 40 + scol) = ha;
      *(v4bf*)(Bs + row * 40 + scol) = hb;
    }
    __syncthreads();
    v16bf a = frag_a_lds(As + wav * 16 * 40, 40, lane);
    v16bf bfr[8];
#pragma unroll
    for (int nt = 0; nt < 8; ++nt) bfr[nt] = frag_b_lds(Bs + nt * 16 * 40, 40, lane);
#pragma unroll
    for (int nt = 0; nt < 8; ++nt) acc[nt] = wmma_bf16(a, bfr[nt], acc[nt]);
  }

  bf16* outb = (bf16*)outp;
  const int half = (lane < 16) ? 0 : 8, col = lane & 15;
#pragma unroll
  for (int nt = 0; nt < 8; ++nt) {
    int n = nBase + nt * 16 + col;
    int h = n >> 6, d = n & 63;
    float bb = bias[n];
#pragma unroll
    for (int r = 0; r < 8; ++r) {
      int m = mBase + wav * 16 + r + half;
      int t = m >> 2, bz = m & 3;
      outb[((size_t)(bz * H_NUM + h) * T_LEN + t) * HD + d] = (bf16)(acc[nt][r] + bb);
    }
  }
}

// ============ Kernel 2: HEPOS-selective K/V projection ============
__global__ void __launch_bounds__(256) proj_kv_kernel(
    const float* __restrict__ key, const float* __restrict__ value,
    const float* __restrict__ Wk, const float* __restrict__ bk,
    const float* __restrict__ Wv, const float* __restrict__ bv,
    unsigned short* __restrict__ kselp, unsigned short* __restrict__ vselTp) {
  __shared__ __align__(16) bf16 As[128 * 40];
  __shared__ __align__(16) bf16 Bs[64 * 40];
  const int tid = threadIdx.x, lane = tid & 31, wav = tid >> 5;
  const int mBase = blockIdx.x * 128;
  const int z = blockIdx.z;
  const int h = z >> 1, which = z & 1;
  const float* X = which ? value : key;
  const float* W = which ? Wv : Wk;
  const float* bias = which ? bv : bk;
  v8f acc[4];
#pragma unroll
  for (int i = 0; i < 4; ++i) acc[i] = zero8();

  const int srow = tid >> 3, scol = (tid & 7) * 4;

  for (int kb = 0; kb < D_DIM; kb += 32) {
    __syncthreads();
    v4f fa[4], fb[2];
#pragma unroll
    for (int it = 0; it < 4; ++it) {   // A: gather HEPOS rows
      int g = mBase + srow + it * 32;
      int j = g >> 2, bz = g & 3;
      int src = (h + STRIDE * j) * B_SZ + bz;
      fa[it] = *(const v4f*)(X + (size_t)src * D_DIM + kb + scol);
    }
#pragma unroll
    for (int it = 0; it < 2; ++it) {   // B: 64-row weight slice
      int row = srow + it * 32;
      fb[it] = *(const v4f*)(W + (size_t)(h * HD + row) * D_DIM + kb + scol);
    }
#pragma unroll
    for (int it = 0; it < 4; ++it) {
      int row = srow + it * 32;
      v4bf ha; ha[0]=(bf16)fa[it][0]; ha[1]=(bf16)fa[it][1]; ha[2]=(bf16)fa[it][2]; ha[3]=(bf16)fa[it][3];
      *(v4bf*)(As + row * 40 + scol) = ha;
    }
#pragma unroll
    for (int it = 0; it < 2; ++it) {
      int row = srow + it * 32;
      v4bf hb; hb[0]=(bf16)fb[it][0]; hb[1]=(bf16)fb[it][1]; hb[2]=(bf16)fb[it][2]; hb[3]=(bf16)fb[it][3];
      *(v4bf*)(Bs + row * 40 + scol) = hb;
    }
    __syncthreads();
    v16bf a = frag_a_lds(As + wav * 16 * 40, 40, lane);
    v16bf bfr[4];
#pragma unroll
    for (int nt = 0; nt < 4; ++nt) bfr[nt] = frag_b_lds(Bs + nt * 16 * 40, 40, lane);
#pragma unroll
    for (int nt = 0; nt < 4; ++nt) acc[nt] = wmma_bf16(a, bfr[nt], acc[nt]);
  }

  bf16* ksel = (bf16*)kselp;
  bf16* vselT = (bf16*)vselTp;
  const int half = (lane < 16) ? 0 : 8, col = lane & 15;
#pragma unroll
  for (int nt = 0; nt < 4; ++nt) {
    int d = nt * 16 + col;
    float bb = bias[h * HD + d];
#pragma unroll
    for (int r = 0; r < 8; ++r) {
      int m = mBase + wav * 16 + r + half;
      int j = m >> 2, bz = m & 3;
      float val = acc[nt][r] + bb;
      if (which == 0)
        ksel[((size_t)(bz * H_NUM + h) * NKV + j) * HD + d] = (bf16)val;
      else
        vselT[((size_t)(bz * H_NUM + h) * HD + d) * NKV + j] = (bf16)val;
    }
  }
}

// ============ Kernel 3: fused flash attention (per b,h / 128-row t tile) ============
__global__ void __launch_bounds__(256) attn_kernel(
    const unsigned short* __restrict__ qp, const unsigned short* __restrict__ kp,
    const unsigned short* __restrict__ vp, unsigned short* __restrict__ op) {
  __shared__ __align__(16) bf16 Qs[128 * 72];   // [t][d]
  __shared__ __align__(16) bf16 Ks[64 * 72];    // [j][d]
  __shared__ __align__(16) bf16 Vs[64 * 72];    // [d][j]  (V transposed)
  __shared__ __align__(16) bf16 Ps[128 * 72];   // [t][j]  attn probs
  const bf16* q_bf = (const bf16*)qp;
  const bf16* ksel = (const bf16*)kp;
  const bf16* vselT = (const bf16*)vp;
  bf16* attn_bf = (bf16*)op;
  const int tid = threadIdx.x, lane = tid & 31, wav = tid >> 5;
  const int tBase = blockIdx.x * 128;
  const int bh = blockIdx.y;          // = bz*16 + h
  const int bz = bh >> 4, h = bh & 15;
  const int m0 = wav * 16;
  const int half = (lane < 16) ? 0 : 8, col = lane & 15;

  // async-load Q tile once (128 x 64 bf16) via ASYNCcnt path
  const size_t qpage = ((size_t)bh * T_LEN + tBase) * HD;
  {
    const int row = tid >> 3, dc = (tid & 7) * 8;     // 8 chunks/row
#pragma unroll
    for (int it = 0; it < 4; ++it)
      async_cp16(Qs + (row + it * 32) * 72 + dc,
                 q_bf + qpage + (size_t)(row + it * 32) * HD + dc);
  }
  async_wait0();
  __syncthreads();
  v16bf a0 = frag_a_lds(Qs + m0 * 72, 72, lane);       // d = 0..31
  v16bf a1 = frag_a_lds(Qs + m0 * 72 + 32, 72, lane);  // d = 32..63

  v8f o[4];
  float mrun[8], lrun[8];
#pragma unroll
  for (int i = 0; i < 4; ++i) o[i] = zero8();
#pragma unroll
  for (int r = 0; r < 8; ++r) { mrun[r] = -1e30f; lrun[r] = 0.f; }

  for (int j0 = 0; j0 < NKV; j0 += 64) {
    __syncthreads();
    // async copy K (64x64 [j][d]) and V (64x64 [d][j]) tiles, one wait
    {
      const int row = tid >> 3, dc = (tid & 7) * 8;
      const size_t kpage = ((size_t)bh * NKV + j0) * HD;
      const size_t vpage = (size_t)bh * HD * NKV + j0;
#pragma unroll
      for (int it = 0; it < 2; ++it)
        async_cp16(Ks + (row + it * 32) * 72 + dc,
                   ksel + kpage + (size_t)(row + it * 32) * HD + dc);
#pragma unroll
      for (int it = 0; it < 2; ++it)
        async_cp16(Vs + (row + it * 32) * 72 + dc,
                   vselT + vpage + (size_t)(row + it * 32) * NKV + dc);
    }
    async_wait0();
    __syncthreads();

    // scores: S = scale * Q K^T  (reduction over d = 64); batch frags, then wmma
    v16bf kb_[8];
#pragma unroll
    for (int jt = 0; jt < 4; ++jt) {
      kb_[2 * jt]     = frag_b_lds(Ks + jt * 16 * 72, 72, lane);
      kb_[2 * jt + 1] = frag_b_lds(Ks + jt * 16 * 72 + 32, 72, lane);
    }
    v8f s[4];
#pragma unroll
    for (int jt = 0; jt < 4; ++jt) {
      v8f c0 = zero8();
      c0 = wmma_bf16(a0, kb_[2 * jt], c0);
      c0 = wmma_bf16(a1, kb_[2 * jt + 1], c0);
#pragma unroll
      for (int r = 0; r < 8; ++r) c0[r] *= 0.125f;   // 64^-0.5
      s[jt] = c0;
    }

    // online softmax: row stats across the 16-lane column group
#pragma unroll
    for (int r = 0; r < 8; ++r) {
      float rm = -1e30f;
#pragma unroll
      for (int jt = 0; jt < 4; ++jt) rm = fmaxf(rm, s[jt][r]);
      rm = fmaxf(rm, __shfl_xor(rm, 1, 32));
      rm = fmaxf(rm, __shfl_xor(rm, 2, 32));
      rm = fmaxf(rm, __shfl_xor(rm, 4, 32));
      rm = fmaxf(rm, __shfl_xor(rm, 8, 32));
      float nm = fmaxf(mrun[r], rm);
      float corr = __expf(mrun[r] - nm);
      float ps = 0.f;
#pragma unroll
      for (int jt = 0; jt < 4; ++jt) {
        float p = __expf(s[jt][r] - nm);
        s[jt][r] = p;
        ps += p;
      }
      ps += __shfl_xor(ps, 1, 32);
      ps += __shfl_xor(ps, 2, 32);
      ps += __shfl_xor(ps, 4, 32);
      ps += __shfl_xor(ps, 8, 32);
      lrun[r] = lrun[r] * corr + ps;
      mrun[r] = nm;
#pragma unroll
      for (int nt = 0; nt < 4; ++nt) o[nt][r] *= corr;
    }

    // prefetch V fragments (independent of P spill)
    v16bf vb_[8];
#pragma unroll
    for (int nt = 0; nt < 4; ++nt) {
      vb_[2 * nt]     = frag_b_lds(Vs + nt * 16 * 72, 72, lane);
      vb_[2 * nt + 1] = frag_b_lds(Vs + nt * 16 * 72 + 32, 72, lane);
    }

    // spill P (own 16 rows) to LDS in A-fragment feed layout
#pragma unroll
    for (int jt = 0; jt < 4; ++jt)
#pragma unroll
      for (int r = 0; r < 8; ++r)
        Ps[(m0 + r + half) * 72 + jt * 16 + col] = (bf16)s[jt][r];

    // O += P V  (reduction over j = 64); wave reads only its own P rows
    v16bf pa0 = frag_a_lds(Ps + m0 * 72, 72, lane);
    v16bf pa1 = frag_a_lds(Ps + m0 * 72 + 32, 72, lane);
#pragma unroll
    for (int nt = 0; nt < 4; ++nt) {
      o[nt] = wmma_bf16(pa0, vb_[2 * nt], o[nt]);
      o[nt] = wmma_bf16(pa1, vb_[2 * nt + 1], o[nt]);
    }
  }

  // epilogue: normalize, store to attn_bf[(t*B+b)][h*64+d]
#pragma unroll
  for (int r = 0; r < 8; ++r) {
    float inv = 1.0f / lrun[r];
    int t = tBase + m0 + r + half;
#pragma unroll
    for (int nt = 0; nt < 4; ++nt) {
      int d = nt * 16 + col;
      attn_bf[(size_t)(t * B_SZ + bz) * D_DIM + h * HD + d] = (bf16)(o[nt][r] * inv);
    }
  }
}

// ============ Kernel 4: output projection ============
__global__ void __launch_bounds__(256) proj_o_kernel(
    const unsigned short* __restrict__ Ap, const float* __restrict__ W,
    const float* __restrict__ bias, float* __restrict__ out) {
  __shared__ __align__(16) bf16 As[128 * 40];
  __shared__ __align__(16) bf16 Bs[128 * 40];
  const bf16* A = (const bf16*)Ap;
  const int tid = threadIdx.x, lane = tid & 31, wav = tid >> 5;
  const int mBase = blockIdx.x * 128;
  const int nBase = blockIdx.y * 128;
  v8f acc[8];
#pragma unroll
  for (int i = 0; i < 8; ++i) acc[i] = zero8();

  const int srow = tid >> 3, scol = (tid & 7) * 4;
  const int arow = tid >> 2, acol = (tid & 3) * 8;     // A is bf16: 4 chunks/row

  for (int kb = 0; kb < D_DIM; kb += 32) {
    __syncthreads();
    v8bf aa[2];
    v4f fb[4];
#pragma unroll
    for (int it = 0; it < 2; ++it)
      aa[it] = *(const v8bf*)(A + (size_t)(mBase + arow + it * 64) * D_DIM + kb + acol);
#pragma unroll
    for (int it = 0; it < 4; ++it)
      fb[it] = *(const v4f*)(W + (size_t)(nBase + srow + it * 32) * D_DIM + kb + scol);
#pragma unroll
    for (int it = 0; it < 2; ++it)
      *(v8bf*)(As + (arow + it * 64) * 40 + acol) = aa[it];
#pragma unroll
    for (int it = 0; it < 4; ++it) {
      v4bf hb; hb[0]=(bf16)fb[it][0]; hb[1]=(bf16)fb[it][1]; hb[2]=(bf16)fb[it][2]; hb[3]=(bf16)fb[it][3];
      *(v4bf*)(Bs + (srow + it * 32) * 40 + scol) = hb;
    }
    __syncthreads();
    v16bf a = frag_a_lds(As + wav * 16 * 40, 40, lane);
    v16bf bfr[8];
#pragma unroll
    for (int nt = 0; nt < 8; ++nt) bfr[nt] = frag_b_lds(Bs + nt * 16 * 40, 40, lane);
#pragma unroll
    for (int nt = 0; nt < 8; ++nt) acc[nt] = wmma_bf16(a, bfr[nt], acc[nt]);
  }

  const int half = (lane < 16) ? 0 : 8, col = lane & 15;
#pragma unroll
  for (int nt = 0; nt < 8; ++nt) {
    int n = nBase + nt * 16 + col;
    float bb = bias[n];
#pragma unroll
    for (int r = 0; r < 8; ++r) {
      int m = mBase + wav * 16 + r + half;
      out[(size_t)m * D_DIM + n] = acc[nt][r] + bb;
    }
  }
}

// ---------------- host launcher ----------------
extern "C" void kernel_launch(void* const* d_in, const int* in_sizes, int n_in,
                              void* d_out, int out_size, void* d_ws, size_t ws_size,
                              hipStream_t stream) {
  const float* query = (const float*)d_in[0];
  const float* key   = (const float*)d_in[1];
  const float* value = (const float*)d_in[2];
  const float* Wq = (const float*)d_in[3];
  const float* bq = (const float*)d_in[4];
  const float* Wk = (const float*)d_in[5];
  const float* bk = (const float*)d_in[6];
  const float* Wv = (const float*)d_in[7];
  const float* bv = (const float*)d_in[8];
  const float* Wo = (const float*)d_in[9];
  const float* bo = (const float*)d_in[10];
  float* out = (float*)d_out;

  char* ws = (char*)d_ws;
  unsigned short* q_bf   = (unsigned short*)(ws);                      // 8 MB [B,H,T,64]
  unsigned short* ksel   = (unsigned short*)(ws + 8u * 1024 * 1024);   // 4 MB [B,H,512,64]
  unsigned short* vselT  = (unsigned short*)(ws + 12u * 1024 * 1024);  // 4 MB [B,H,64,512]
  unsigned short* attnbf = (unsigned short*)(ws + 16u * 1024 * 1024);  // 8 MB [T*B, D]

  proj_q_kernel<<<dim3(32, 8), 256, 0, stream>>>(query, Wq, bq, q_bf);
  proj_kv_kernel<<<dim3(16, 1, 32), 256, 0, stream>>>(key, value, Wk, bk, Wv, bv, ksel, vselT);
  attn_kernel<<<dim3(8, 64), 256, 0, stream>>>(q_bf, ksel, vselT, attnbf);
  proj_o_kernel<<<dim3(32, 8), 256, 0, stream>>>(attnbf, Wo, bo, out);
}
